// DeformableBlock_13932873909314
// MI455X (gfx1250) — matrix operand order repaired and checked by reading, not compiled, not run on hardware
//
#include <hip/hip_runtime.h>
#include <cmath>

// ---------------- problem constants ----------------
#define BB   4
#define CC   64
#define HH   128
#define WW   128
#define OO   64
#define KS   3
#define KK   9                 // KS*KS
#define KTOT (CC * KK)         // 576 = reduction length (K index = kk*64 + c)
#define PIX  64                // pixels per workgroup tile (N tile)
#define BK   (KTOT + 8)        // LDS row stride in halfs (pad, keeps 16B align: 584*2=1168)
#define KSTEP 32
#define NSTEP (KTOT / KSTEP)   // 18

typedef _Float16 h_t;
typedef __attribute__((ext_vector_type(8)))  _Float16 v8h;
typedef __attribute__((ext_vector_type(16))) _Float16 v16h;
typedef __attribute__((ext_vector_type(8)))  float    v8f;

static __device__ inline v8h splat8(float f) {
    _Float16 v = (_Float16)f;
    v8h r = {v, v, v, v, v, v, v, v};
    return r;
}

// A-fragment (16x32 f16, M x K) per ISA layout:
//   lane l: row M = l&15 ; lh = l>>4
//   VGPR0..3 = K[lh*8 .. lh*8+8) ; VGPR4..7 = K[16+lh*8 .. 16+lh*8+8)
// wp is row-major [rows][KTOT] f16.
static __device__ inline v16h load_a_frag(const h_t* __restrict__ wp,
                                          int mbase, int kbase, int lane) {
    int row = mbase + (lane & 15);
    int lh  = lane >> 4;
    const v8h* p1 = (const v8h*)(wp + row * KTOT + kbase + lh * 8);
    const v8h* p2 = (const v8h*)(wp + row * KTOT + kbase + lh * 8 + 16);
    v8h a = *p1, b = *p2;
    v16h r;
#pragma unroll
    for (int i = 0; i < 8; ++i) { r[i] = a[i]; r[i + 8] = b[i]; }
    return r;
}

// B-fragment (32x16 f16, K x N) per ISA striping:
//   lane l: col N = l&15 ; K[(l>>4)*8 .. +8) then K[16+(l>>4)*8 .. +8)
// Bt is [pixel][BK] f16 (K contiguous per pixel).
static __device__ inline v16h load_b_frag(const h_t* Bt, int nbase, int kbase, int lane) {
    const v8h* p = (const v8h*)(Bt + (nbase + (lane & 15)) * BK + kbase + (lane >> 4) * 8);
    v8h a = p[0], b = p[2];
    v16h r;
#pragma unroll
    for (int i = 0; i < 8; ++i) { r[i] = a[i]; r[i + 8] = b[i]; }
    return r;
}

// ---------------- kernel T: NCHW f32 -> NHWC f16 ----------------
__global__ void k_nchw_to_nhwc(const float* __restrict__ x, h_t* __restrict__ xh) {
    __shared__ float tile[32][33];
    int bh  = blockIdx.x;            // b*H + h
    int b   = bh / HH, h = bh % HH;
    int x0  = blockIdx.y * 32;
    int c0  = blockIdx.z * 32;
    int tid = threadIdx.x;
#pragma unroll
    for (int i = 0; i < 4; ++i) {
        int idx = tid + i * 256;
        int cc = idx >> 5, xx = idx & 31;
        tile[cc][xx] = x[((b * CC + c0 + cc) * HH + h) * WW + x0 + xx];
    }
    __syncthreads();
#pragma unroll
    for (int i = 0; i < 4; ++i) {
        int idx = tid + i * 256;
        int xx = idx >> 5, cc = idx & 31;
        xh[((b * HH + h) * WW + x0 + xx) * CC + c0 + cc] = (h_t)tile[cc][xx];
    }
}

// ---------------- kernel P: pack weights to f16, K index = kk*64 + c ----------------
__global__ void k_pack_weights(const float* __restrict__ w_off, const float* __restrict__ w_def,
                               h_t* __restrict__ wOffP, h_t* __restrict__ wDefP) {
    int i = blockIdx.x * 256 + threadIdx.x;
    const int N1 = 32 * KTOT;     // offset-conv rows padded 18 -> 32
    const int N2 = OO * KTOT;
    if (i < N1) {
        int o = i / KTOT, k = i % KTOT, kk = k >> 6, c = k & 63;
        float v = (o < 18) ? w_off[(o * CC + c) * KK + kk] : 0.0f;
        wOffP[i] = (h_t)v;
    } else if (i < N1 + N2) {
        int j = i - N1;
        int o = j / KTOT, k = j % KTOT, kk = k >> 6, c = k & 63;
        wDefP[j] = (h_t)w_def[(o * CC + c) * KK + kk];
    }
}

// ---------------- kernel C: offset conv (im2col + WMMA), Hardtanh, store NHWC ----------------
__global__ void k_offset_gemm(const h_t* __restrict__ xh, const h_t* __restrict__ wOffP,
                              const float* __restrict__ b_off, float* __restrict__ off) {
    extern __shared__ char smem[];
    h_t* Bt = (h_t*)smem;                       // [PIX][BK]
    int blk = blockIdx.x;
    const int tpr = WW / PIX;                   // tiles per row
    int b   = blk / (HH * tpr);
    int rem = blk % (HH * tpr);
    int h   = rem / tpr;
    int x0  = (rem % tpr) * PIX;
    int tid = threadIdx.x;

    // build im2col patch tile: Bt[p][kk*64+c] = x[b, h+ky, x+kx, c] (zero pad)
    for (int pair = tid; pair < PIX * KK; pair += 256) {
        int p = pair / KK, kk = pair % KK;
        int y  = h + (kk / 3) - 1;
        int xg = x0 + p + (kk % 3) - 1;
        v8h* dst = (v8h*)(Bt + p * BK + kk * CC);
        if (y >= 0 && y < HH && xg >= 0 && xg < WW) {
            const v8h* src = (const v8h*)(xh + ((b * HH + y) * WW + xg) * CC);
#pragma unroll
            for (int i = 0; i < 8; ++i) dst[i] = src[i];
        } else {
            v8h z = {};
#pragma unroll
            for (int i = 0; i < 8; ++i) dst[i] = z;
        }
    }
    __syncthreads();

    int lane = tid & 31, wave = tid >> 5;       // 8 wave32s
    int nblk = wave & 3, mblk = wave >> 2;      // 2(M) x 4(N) tiles, one per wave
    v8f acc = {};
    for (int s = 0; s < NSTEP; ++s) {
        v16h bf = load_b_frag(Bt, nblk * 16, s * KSTEP, lane);
        v16h af = load_a_frag(wOffP, mblk * 16, s * KSTEP, lane);
        acc = __builtin_amdgcn_wmma_f32_16x16x32_f16(false, af, false, bf,
                                                     (short)0, acc, false, false);
    }
    int pxg = x0 + nblk * 16 + (lane & 15);
    int lh  = lane >> 4;
#pragma unroll
    for (int v = 0; v < 8; ++v) {
        int o = mblk * 16 + v + 8 * lh;
        if (o < 18) {
            float r = acc[v] + b_off[o];
            r = fminf(1.0f, fmaxf(-1.0f, r));   // Hardtanh
            off[((b * HH + h) * WW + pxg) * 18 + o] = r;
        }
    }
}

// ---------------- kernel D: deformable conv (bilinear sample -> WMMA) ----------------
__global__ void k_deform_gemm(const h_t* __restrict__ xh, const float* __restrict__ off,
                              const h_t* __restrict__ wDefP, const float* __restrict__ b_def,
                              float* __restrict__ out) {
    extern __shared__ char smem[];
    h_t* Bt = (h_t*)smem;                       // [PIX][BK]
    int blk = blockIdx.x;
    const int tpr = WW / PIX;
    int b   = blk / (HH * tpr);
    int rem = blk % (HH * tpr);
    int h   = rem / tpr;
    int x0  = (rem % tpr) * PIX;
    int tid = threadIdx.x;

    // phase 1: bilinear sampling, Bt[p][kk*64+c]
    for (int pair = tid; pair < PIX * KK; pair += 256) {
        int p = pair / KK, kk = pair % KK;
        int xx = x0 + p;
        const float* o2 = off + ((b * HH + h) * WW + xx) * 18 + kk * 2;
        float dy = o2[0], dx = o2[1];
        float py = (float)(h + (kk / 3) - 1) + dy;
        float px = (float)(xx + (kk % 3) - 1) + dx;
        float fy = floorf(py), fx = floorf(px);
        float ly = py - fy, lx = px - fx;
        int yi = (int)fy, xi = (int)fx;
        float w00 = (1.f - ly) * (1.f - lx);
        float w01 = (1.f - ly) * lx;
        float w10 = ly * (1.f - lx);
        float w11 = ly * lx;
        bool vy0 = (yi >= 0) && (yi < HH);
        bool vy1 = (yi + 1 >= 0) && (yi + 1 < HH);
        bool vx0 = (xi >= 0) && (xi < WW);
        bool vx1 = (xi + 1 >= 0) && (xi + 1 < WW);
        if (!(vy0 && vx0)) w00 = 0.f;
        if (!(vy0 && vx1)) w01 = 0.f;
        if (!(vy1 && vx0)) w10 = 0.f;
        if (!(vy1 && vx1)) w11 = 0.f;
        int y0c = min(max(yi, 0), HH - 1), y1c = min(max(yi + 1, 0), HH - 1);
        int x0c = min(max(xi, 0), WW - 1), x1c = min(max(xi + 1, 0), WW - 1);
        const v8h* q00 = (const v8h*)(xh + ((b * HH + y0c) * WW + x0c) * CC);
        const v8h* q01 = (const v8h*)(xh + ((b * HH + y0c) * WW + x1c) * CC);
        const v8h* q10 = (const v8h*)(xh + ((b * HH + y1c) * WW + x0c) * CC);
        const v8h* q11 = (const v8h*)(xh + ((b * HH + y1c) * WW + x1c) * CC);
        v8h W00 = splat8(w00), W01 = splat8(w01), W10 = splat8(w10), W11 = splat8(w11);
        v8h* dst = (v8h*)(Bt + p * BK + kk * CC);
#pragma unroll
        for (int i = 0; i < 8; ++i)
            dst[i] = q00[i] * W00 + q01[i] * W01 + q10[i] * W10 + q11[i] * W11;
    }
    __syncthreads();

    // phase 2: 8 waves; wave handles N-block (wave&3) and M-blocks {2*(wave>>2), +1}
    int lane = tid & 31, wave = tid >> 5;
    int nblk = wave & 3;
    int m0   = (wave >> 2) * 2;
    v8f acc0 = {}, acc1 = {};
    for (int s = 0; s < NSTEP; ++s) {
        v16h bf = load_b_frag(Bt, nblk * 16, s * KSTEP, lane);
        v16h a0 = load_a_frag(wDefP, (m0 + 0) * 16, s * KSTEP, lane);
        v16h a1 = load_a_frag(wDefP, (m0 + 1) * 16, s * KSTEP, lane);
        acc0 = __builtin_amdgcn_wmma_f32_16x16x32_f16(false, a0, false, bf,
                                                      (short)0, acc0, false, false);
        acc1 = __builtin_amdgcn_wmma_f32_16x16x32_f16(false, a1, false, bf,
                                                      (short)0, acc1, false, false);
    }
    int pxg = x0 + nblk * 16 + (lane & 15);
    int lh  = lane >> 4;
#pragma unroll
    for (int v = 0; v < 8; ++v) {
        int o0 = (m0 + 0) * 16 + v + 8 * lh;
        int o1 = (m0 + 1) * 16 + v + 8 * lh;
        out[((b * OO + o0) * HH + h) * WW + pxg] = acc0[v] + b_def[o0];
        out[((b * OO + o1) * HH + h) * WW + pxg] = acc1[v] + b_def[o1];
    }
}

// ---------------- host launcher ----------------
extern "C" void kernel_launch(void* const* d_in, const int* in_sizes, int n_in,
                              void* d_out, int out_size, void* d_ws, size_t ws_size,
                              hipStream_t stream) {
    const float* x     = (const float*)d_in[0];
    const float* w_off = (const float*)d_in[1];
    const float* b_off = (const float*)d_in[2];
    const float* w_def = (const float*)d_in[3];
    const float* b_def = (const float*)d_in[4];
    float* out = (float*)d_out;

    // workspace carve (all 256B aligned)
    char* ws = (char*)d_ws;
    const size_t szXh  = (size_t)BB * HH * WW * CC * sizeof(h_t);   // 8,388,608
    const size_t szOff = (size_t)BB * HH * WW * 18 * sizeof(float); // 4,718,592
    const size_t szWo  = (size_t)32 * KTOT * sizeof(h_t);           // 36,864
    h_t*   xh    = (h_t*)ws;
    float* offb  = (float*)(ws + szXh);
    h_t*   wOffP = (h_t*)(ws + szXh + szOff);
    h_t*   wDefP = (h_t*)(ws + szXh + szOff + szWo);

    // T: layout/precision prep
    dim3 gT(BB * HH, WW / 32, CC / 32);
    k_nchw_to_nhwc<<<gT, 256, 0, stream>>>(x, xh);

    // P: weight packing
    int nPack = 32 * KTOT + OO * KTOT;
    k_pack_weights<<<(nPack + 255) / 256, 256, 0, stream>>>(w_off, w_def, wOffP, wDefP);

    // dynamic LDS for the B tiles (74,752 B < 320 KB/WGP, > default 64 KB cap)
    size_t ldsBytes = (size_t)PIX * BK * sizeof(h_t);
    hipFuncSetAttribute((const void*)k_offset_gemm,
                        hipFuncAttributeMaxDynamicSharedMemorySize, (int)ldsBytes);
    hipFuncSetAttribute((const void*)k_deform_gemm,
                        hipFuncAttributeMaxDynamicSharedMemorySize, (int)ldsBytes);

    int nBlk = BB * HH * (WW / PIX);  // 1024 workgroups
    k_offset_gemm<<<nBlk, 256, ldsBytes, stream>>>(xh, wOffP, b_off, offb);
    k_deform_gemm<<<nBlk, 256, ldsBytes, stream>>>(xh, offb, wDefP, b_def, out);
}
